// Attention_81157702025384
// MI455X (gfx1250) — compile-verified, matrix operand order checked by reading, not run on hardware
//
#include <hip/hip_runtime.h>

// ---------------------------------------------------------------------------
// Problem constants (B=4, S=2048, D=2048) — all tile-divisible.
// ---------------------------------------------------------------------------
#define B_ 4
#define S_ 2048
#define D_ 2048

typedef __attribute__((ext_vector_type(16))) __bf16 bf16x16;
typedef __attribute__((ext_vector_type(8)))  float  f32x8;
typedef __attribute__((ext_vector_type(8)))  unsigned short us8;
typedef __attribute__((ext_vector_type(4)))  int    i32x4;

union Frag {
    bf16x16 v;
    us8     h[2];
};

__device__ __forceinline__ unsigned short f2b(float f) {
    unsigned int u = __float_as_uint(f);
    unsigned int r = u + 0x7FFFu + ((u >> 16) & 1u);  // round-to-nearest-even
    return (unsigned short)(r >> 16);
}
__device__ __forceinline__ float b2f(unsigned short h) {
    return __uint_as_float(((unsigned int)h) << 16);
}

// ---- CDNA5 async global->LDS copy (ASYNCcnt path), guarded ----------------
#if defined(__has_builtin)
#if __has_builtin(__builtin_amdgcn_global_load_async_to_lds_b128)
#define USE_ASYNC_LDS 1
#endif
#endif
#ifndef USE_ASYNC_LDS
#define USE_ASYNC_LDS 0
#endif

#define AS1_P4(p) ((__attribute__((address_space(1))) i32x4*)(p))
#define AS3_P4(p) ((__attribute__((address_space(3))) i32x4*)(p))

template <int N>
__device__ __forceinline__ void wait_async_le() {
#if USE_ASYNC_LDS
#if __has_builtin(__builtin_amdgcn_s_wait_asynccnt)
    __builtin_amdgcn_s_wait_asynccnt(N);  // immediate required
#else
    asm volatile("s_wait_asynccnt %0" ::"i"(N) : "memory");
#endif
#endif
}

// ---------------------------------------------------------------------------
// f32 -> bf16 conversion (4 elements / thread)
// ---------------------------------------------------------------------------
__global__ __launch_bounds__(256) void cvt_f32_bf16(const float* __restrict__ in,
                                                    unsigned short* __restrict__ out,
                                                    int n) {
    int i = (blockIdx.x * 256 + threadIdx.x) * 4;
    if (i + 3 < n) {
        float4 v = *(const float4*)(in + i);
        out[i + 0] = f2b(v.x);
        out[i + 1] = f2b(v.y);
        out[i + 2] = f2b(v.z);
        out[i + 3] = f2b(v.w);
    }
}

// ---------------------------------------------------------------------------
// Generic NT bf16 WMMA GEMM:  C[m,n] = sum_k A[m,k] * B[n,k]
//   Block tile 128x128, BK=32, triple-buffered LDS fed by async global->LDS
//   (stage s+2 issued each iteration; s_wait_asynccnt<=4 => stage s landed).
//   8 waves; each wave owns 32(M) x 64(N) = 2x4 WMMA accumulators.
//   MODE 0: store bf16, MODE 1: store f32.
// ---------------------------------------------------------------------------
#define BM 128
#define BN 128
#define BK 32
#define LDSK 40  // padded row stride (halfs) to spread LDS banks

template <int MODE>
__global__ __launch_bounds__(256) void gemm_nt_bf16(
    const unsigned short* __restrict__ A, size_t batchA,
    const unsigned short* __restrict__ Bm, size_t batchB,
    void* __restrict__ Cv, size_t batchC,
    int K, int lda, int ldb, int ldc) {
    const int bz = blockIdx.z;
    A  += (size_t)bz * batchA;
    Bm += (size_t)bz * batchB;

    const int m0 = blockIdx.y * BM;
    const int n0 = blockIdx.x * BN;

    __shared__ unsigned short sA[3][BM * LDSK] __attribute__((aligned(16)));
    __shared__ unsigned short sB[3][BN * LDSK] __attribute__((aligned(16)));

    const int t    = threadIdx.x;
    const int lane = t & 31;
    const int wid  = t >> 5;   // 0..7
    const int wm   = wid & 3;  // 0..3 -> 32 rows each
    const int wn   = wid >> 2; // 0..1 -> 64 cols each

    // Staging assignment: thread t copies one 32B slice of A and one of B:
    //   row = t>>1 (0..127), col = (t&1)*16 halfs, as two b128 chunks.
    const int srow = t >> 1;
    const int scol = (t & 1) * 16;
    const unsigned short* gA = A  + (size_t)(m0 + srow) * lda + scol;
    const unsigned short* gB = Bm + (size_t)(n0 + srow) * ldb + scol;
    const int lofs = srow * LDSK + scol;

#if USE_ASYNC_LDS
    auto stage = [&](int buf, int k0) {
        __builtin_amdgcn_global_load_async_to_lds_b128(AS1_P4(gA + k0),
                                                       AS3_P4(&sA[buf][lofs]), 0, 0);
        __builtin_amdgcn_global_load_async_to_lds_b128(AS1_P4(gA + k0 + 8),
                                                       AS3_P4(&sA[buf][lofs + 8]), 0, 0);
        __builtin_amdgcn_global_load_async_to_lds_b128(AS1_P4(gB + k0),
                                                       AS3_P4(&sB[buf][lofs]), 0, 0);
        __builtin_amdgcn_global_load_async_to_lds_b128(AS1_P4(gB + k0 + 8),
                                                       AS3_P4(&sB[buf][lofs + 8]), 0, 0);
    };
#else
    us8 pa0, pa1, pb0, pb1;
    auto fetch = [&](int k0) {
        pa0 = *(const us8*)(gA + k0);
        pa1 = *(const us8*)(gA + k0 + 8);
        pb0 = *(const us8*)(gB + k0);
        pb1 = *(const us8*)(gB + k0 + 8);
    };
    auto commit = [&](int buf) {
        *(us8*)&sA[buf][lofs]     = pa0;
        *(us8*)&sA[buf][lofs + 8] = pa1;
        *(us8*)&sB[buf][lofs]     = pb0;
        *(us8*)&sB[buf][lofs + 8] = pb1;
    };
#endif

    f32x8 acc[2][4] = {};

    const int kc     = (lane >> 4) * 8;  // per-lane K chunk select (ISA layout)
    const int nsteps = K / BK;

    // Fragment compute from LDS buffer `cur`.
    auto compute = [&](int cur) {
        Frag fa[2], fb[4];
#pragma unroll
        for (int im = 0; im < 2; ++im) {
            int r = wm * 32 + im * 16 + (lane & 15);
            fa[im].h[0] = *(const us8*)&sA[cur][r * LDSK + kc];
            fa[im].h[1] = *(const us8*)&sA[cur][r * LDSK + kc + 16];
        }
#pragma unroll
        for (int in = 0; in < 4; ++in) {
            int r = wn * 64 + in * 16 + (lane & 15);
            fb[in].h[0] = *(const us8*)&sB[cur][r * LDSK + kc];
            fb[in].h[1] = *(const us8*)&sB[cur][r * LDSK + kc + 16];
        }
#pragma unroll
        for (int im = 0; im < 2; ++im)
#pragma unroll
            for (int in = 0; in < 4; ++in)
                acc[im][in] = __builtin_amdgcn_wmma_f32_16x16x32_bf16(
                    false, fa[im].v, false, fb[in].v,
                    (short)0, acc[im][in], false, false);
    };

#if USE_ASYNC_LDS
    // Triple-buffered pipeline: two stages in flight at all times.
    stage(0, 0);
    if (nsteps > 1) stage(1, BK);
    int cur = 0, nx2 = 2;
    for (int s = 0; s < nsteps; ++s) {
        wait_async_le<4>();  // stage s landed (stage s+1 may still be in flight)
        __syncthreads();     // all waves' stage-s writes visible; buf (s-1)%3 free
        if (s + 2 < nsteps) stage(nx2, (s + 2) * BK);  // fire-and-forget
        compute(cur);
        cur = (cur == 2) ? 0 : cur + 1;
        nx2 = (nx2 == 2) ? 0 : nx2 + 1;
    }
#else
    // Fallback: double-buffered, software-pipelined plain loads.
    fetch(0);
    commit(0);
    __syncthreads();
    for (int s = 0; s < nsteps; ++s) {
        const int cur = s & 1;
        if (s + 1 < nsteps) fetch((s + 1) * BK);
        compute(cur);
        if (s + 1 < nsteps) commit(cur ^ 1);
        __syncthreads();
    }
#endif

    // ---- epilogue: C layout VGPR i -> M = i + 8*(lane>=16), N = lane&15 ----
#pragma unroll
    for (int im = 0; im < 2; ++im)
#pragma unroll
        for (int in = 0; in < 4; ++in) {
            int rbase = m0 + wm * 32 + im * 16 + ((lane >> 4) * 8);
            int c     = n0 + wn * 64 + in * 16 + (lane & 15);
            if (MODE == 0) {
                unsigned short* C = (unsigned short*)Cv + (size_t)bz * batchC;
#pragma unroll
                for (int i = 0; i < 8; ++i)
                    C[(size_t)(rbase + i) * ldc + c] = f2b(acc[im][in][i]);
            } else {
                float* C = (float*)Cv + (size_t)bz * batchC;
#pragma unroll
                for (int i = 0; i < 8; ++i)
                    C[(size_t)(rbase + i) * ldc + c] = acc[im][in][i];
            }
        }
}

// ---------------------------------------------------------------------------
// RoPE (HF style) applied in-place to bf16 Q and K.
// ---------------------------------------------------------------------------
__global__ __launch_bounds__(256) void rope_qk(unsigned short* __restrict__ q,
                                               unsigned short* __restrict__ k,
                                               const float* __restrict__ cosT,
                                               const float* __restrict__ sinT) {
    const int HALF = D_ / 2;
    size_t i = (size_t)blockIdx.x * 256 + threadIdx.x;  // over B*S*HALF
    int d     = (int)(i % HALF);
    size_t bs = i / HALF;
    int s     = (int)(bs % S_);
    size_t base = bs * (size_t)D_;

    float c1 = cosT[(size_t)s * D_ + d];
    float s1 = sinT[(size_t)s * D_ + d];
    float c2 = cosT[(size_t)s * D_ + d + HALF];
    float s2 = sinT[(size_t)s * D_ + d + HALF];

    float q1 = b2f(q[base + d]), q2 = b2f(q[base + d + HALF]);
    q[base + d]        = f2b(q1 * c1 - q2 * s1);
    q[base + d + HALF] = f2b(q2 * c2 + q1 * s2);

    float k1 = b2f(k[base + d]), k2 = b2f(k[base + d + HALF]);
    k[base + d]        = f2b(k1 * c1 - k2 * s1);
    k[base + d + HALF] = f2b(k2 * c2 + k1 * s2);
}

// ---------------------------------------------------------------------------
// Softmax with the reference's quirks:
//   masked (j > r) logits stay 0 (multiplicative mask), scale = 1/sqrt(S).
// ---------------------------------------------------------------------------
__global__ __launch_bounds__(256) void softmax_rows(const float* __restrict__ sc,
                                                    unsigned short* __restrict__ p) {
    const int row = blockIdx.x;  // 0 .. B*S-1
    const int r   = row % S_;
    const float* srow = sc + (size_t)row * S_;
    unsigned short* prow = p + (size_t)row * S_;
    const float inv = 0.022097086912079608f;  // 1/sqrt(2048)

    __shared__ float red[256];
    const int t = threadIdx.x;

    float lmax = -1e30f;
    for (int j = t; j < S_; j += 256) {
        float v = (j <= r) ? srow[j] * inv : 0.0f;
        lmax = fmaxf(lmax, v);
    }
    red[t] = lmax;
    __syncthreads();
    for (int o = 128; o > 0; o >>= 1) {
        if (t < o) red[t] = fmaxf(red[t], red[t + o]);
        __syncthreads();
    }
    float m = red[0];
    __syncthreads();

    float lsum = 0.0f;
    for (int j = t; j < S_; j += 256) {
        float v = (j <= r) ? srow[j] * inv : 0.0f;
        lsum += __expf(v - m);
    }
    red[t] = lsum;
    __syncthreads();
    for (int o = 128; o > 0; o >>= 1) {
        if (t < o) red[t] += red[t + o];
        __syncthreads();
    }
    float rs = 1.0f / red[0];

    for (int j = t; j < S_; j += 256) {
        float v = (j <= r) ? srow[j] * inv : 0.0f;
        prow[j] = f2b(__expf(v - m) * rs);
    }
}

// ---------------------------------------------------------------------------
// Orchestration
// ---------------------------------------------------------------------------
extern "C" void kernel_launch(void* const* d_in, const int* in_sizes, int n_in,
                              void* d_out, int out_size, void* d_ws, size_t ws_size,
                              hipStream_t stream) {
    const float* x    = (const float*)d_in[0];  // [B,S,D]
    const float* cosT = (const float*)d_in[1];  // [S,D]
    const float* sinT = (const float*)d_in[2];  // [S,D]
    const float* Wq   = (const float*)d_in[3];  // [D,D]
    const float* Wk   = (const float*)d_in[4];
    const float* Wv   = (const float*)d_in[5];

    const size_t BSD = (size_t)B_ * S_ * D_;  // 16,777,216
    const size_t DD  = (size_t)D_ * D_;       //  4,194,304
    const size_t BSS = (size_t)B_ * S_ * S_;  // 16,777,216

    char* w = (char*)d_ws;
    unsigned short* xb  = (unsigned short*)w; w += BSD * 2;
    unsigned short* wqb = (unsigned short*)w; w += DD * 2;
    unsigned short* wkb = (unsigned short*)w; w += DD * 2;
    unsigned short* wvb = (unsigned short*)w; w += DD * 2;
    unsigned short* qb  = (unsigned short*)w; w += BSD * 2;
    unsigned short* kb  = (unsigned short*)w; w += BSD * 2;
    unsigned short* vtb = (unsigned short*)w; w += BSD * 2;  // per batch [D,S]
    float*          sc  = (float*)w;          w += BSS * 4;
    unsigned short* pb  = (unsigned short*)w; w += BSS * 2;

    // 1) convert inputs to bf16
    cvt_f32_bf16<<<(unsigned)(BSD / 1024), 256, 0, stream>>>(x, xb, (int)BSD);
    cvt_f32_bf16<<<(unsigned)(DD / 1024), 256, 0, stream>>>(Wq, wqb, (int)DD);
    cvt_f32_bf16<<<(unsigned)(DD / 1024), 256, 0, stream>>>(Wk, wkb, (int)DD);
    cvt_f32_bf16<<<(unsigned)(DD / 1024), 256, 0, stream>>>(Wv, wvb, (int)DD);

    dim3 blk(256);

    // 2) Q = x @ Wq^T   (M=S, N=D, K=D)  -> bf16
    gemm_nt_bf16<0><<<dim3(D_ / BN, S_ / BM, B_), blk, 0, stream>>>(
        xb, (size_t)S_ * D_, wqb, 0, qb, (size_t)S_ * D_, D_, D_, D_, D_);
    //    K = x @ Wk^T
    gemm_nt_bf16<0><<<dim3(D_ / BN, S_ / BM, B_), blk, 0, stream>>>(
        xb, (size_t)S_ * D_, wkb, 0, kb, (size_t)S_ * D_, D_, D_, D_, D_);
    //    V^T = Wv @ x^T  (M=D, N=S, K=D) -> bf16, stored [D,S] per batch
    gemm_nt_bf16<0><<<dim3(S_ / BN, D_ / BM, B_), blk, 0, stream>>>(
        wvb, 0, xb, (size_t)S_ * D_, vtb, (size_t)D_ * S_, D_, D_, D_, S_);

    // 3) RoPE in-place on Q, K
    rope_qk<<<(unsigned)((size_t)B_ * S_ * (D_ / 2) / 256), 256, 0, stream>>>(
        qb, kb, cosT, sinT);

    // 4) scores = Q @ K^T (M=S, N=S, K=D) -> f32
    gemm_nt_bf16<1><<<dim3(S_ / BN, S_ / BM, B_), blk, 0, stream>>>(
        qb, (size_t)S_ * D_, kb, (size_t)S_ * D_, sc, (size_t)S_ * S_, D_, D_, D_, S_);

    // 5) softmax with multiplicative causal mask, 1/sqrt(S) scale -> bf16 P
    softmax_rows<<<B_ * S_, 256, 0, stream>>>(sc, pb);

    // 6) out = P @ V  via A=P rows, B=V^T rows (M=S, N=D, K=S) -> f32 d_out
    gemm_nt_bf16<1><<<dim3(D_ / BN, S_ / BM, B_), blk, 0, stream>>>(
        pb, (size_t)S_ * S_, vtb, (size_t)D_ * S_, d_out, (size_t)S_ * D_, S_, S_, S_, D_);
}
